// LiquidNeuralNetwork_73632919322793
// MI455X (gfx1250) — matrix operand order repaired and checked by reading, not compiled, not run on hardware
//
#include <hip/hip_runtime.h>

// ---------------------------------------------------------------------------
// Liquid NN on CDNA5 (gfx1250).
//  - One-time f32->bf16 conversion of x / Wi / L (v_cvt_pk_bf16_f32 when avail)
//  - All inter-layer tensors kept bf16; WMMA f32 accumulate everywhere.
//  - GEMM staging = pure async global->LDS b128 copies with per-lane permuted
//    LDS destinations (fragment order), so v_wmma A reads are contiguous 32B; B is gathered.
// ---------------------------------------------------------------------------

#define BATCH 4096
#define SDIM  64
#define OUTD  12

typedef __attribute__((ext_vector_type(16))) __bf16 v16bf;
typedef __attribute__((ext_vector_type(8)))  float  v8f;

// ---------------- feature probes -------------------------------------------
#if __has_builtin(__builtin_amdgcn_global_load_async_to_lds_b128)
#define HAVE_ASYNC 1
// Parameter type per hipcc diagnostic: generic pointer to 16B int vector.
typedef int v4i_ld __attribute__((vector_size(16)));
#else
#define HAVE_ASYNC 0
#endif

// ---------------- bf16 helpers ---------------------------------------------
static __device__ __forceinline__ unsigned short f2bf(float f) {
  unsigned u = __float_as_uint(f);
  u += 0x7FFFu + ((u >> 16) & 1u);          // round-to-nearest-even
  return (unsigned short)(u >> 16);
}
static __device__ __forceinline__ float bf2f(unsigned short h) {
  return __uint_as_float((unsigned)h << 16);
}
static __device__ __forceinline__ unsigned cvt2(float a, float b) {
#if __has_builtin(__builtin_amdgcn_cvt_pk_bf16_f32)
  auto p = __builtin_amdgcn_cvt_pk_bf16_f32(a, b);   // v_cvt_pk_bf16_f32
  return __builtin_bit_cast(unsigned, p);
#else
  return (unsigned)f2bf(a) | ((unsigned)f2bf(b) << 16);
#endif
}

// 16-byte global(bf16) -> LDS copy; async path bypasses VGPRs (ASYNCcnt).
static __device__ __forceinline__ void cp16(const unsigned short* g,
                                            unsigned short* l) {
#if HAVE_ASYNC
  __builtin_amdgcn_global_load_async_to_lds_b128((v4i_ld*)g, (v4i_ld*)l, 0, 0);
#else
  *(uint4*)l = *(const uint4*)g;
#endif
}
static __device__ __forceinline__ void wait_cp() {
#if HAVE_ASYNC
#if __has_builtin(__builtin_amdgcn_s_wait_asynccnt)
  __builtin_amdgcn_s_wait_asynccnt(0);
#else
  asm volatile("s_wait_asynccnt 0x0" ::: "memory");
#endif
#endif
}

static __device__ __forceinline__ float liquid_act(float x) {
  return tanhf(x) + 0.1f * sinf(0.5f * x) * cosf(0.3f * x);
}

// Fragment-order LDS offsets (elements). 8-element granules are contiguous.
// A-matrix (16x32 bf16, ISA 7.12.2): r = tile row (0..63), c8 = 8-aligned k.
static __device__ __forceinline__ int a_frag_off(int r, int c8) {
  int kk = c8 & 31;
  return ((((r >> 4) << 1) + (c8 >> 5)) * 32 +
          ((r & 15) + (((kk >> 3) & 1) << 4))) * 16 + ((kk >> 4) << 3);
}
// B-matrix (32x16 bf16): r = k row (0..63), c8 = 8-aligned column.
static __device__ __forceinline__ int b_frag_off(int r, int c8) {
  return ((((r >> 5) << 2) + (c8 >> 4)) * 32 + (r & 31)) * 16 + (c8 & 15);
}

// ---------------------------------------------------------------------------
// Kernel 0: elementwise f32 -> bf16 (n divisible by 4)
// ---------------------------------------------------------------------------
__global__ void cvt_bf16_kernel(const float* __restrict__ src,
                                unsigned short* __restrict__ dst, int n4) {
  int i = blockIdx.x * blockDim.x + threadIdx.x;
  if (i >= n4) return;
  float4 v = ((const float4*)src)[i];
  ((uint2*)dst)[i] = make_uint2(cvt2(v.x, v.y), cvt2(v.z, v.w));
}

// ---------------------------------------------------------------------------
// Kernel 1: per-neuron fused GEMM + bias + liquid activation + Wo reduction.
//   grid = (B/64, N), block = 256 (8 wave32). C tile 64x64 per block.
// ---------------------------------------------------------------------------
__global__ __launch_bounds__(256)
void liquid_layer_kernel(const unsigned short* __restrict__ X,   // [B,K] bf16
                         const unsigned short* __restrict__ Wi,  // [N,K,S] bf16
                         const float* __restrict__ bi,           // [N,S]
                         const float* __restrict__ bl,           // [N,S]
                         const float* __restrict__ Wo,           // [N,S]
                         const float* __restrict__ bo,           // [N]
                         unsigned short* __restrict__ P,         // [B,N] bf16
                         int K, int Nn)
{
  __shared__ __align__(32) unsigned short lA[4096];
  __shared__ __align__(32) unsigned short lB[4096];
  __shared__ float lred[64];

  const int t    = threadIdx.x;
  const int bb   = blockIdx.x * 64;
  const int n    = blockIdx.y;
  const int w    = t >> 5;
  const int lane = t & 31;
  const int mblk = w & 3;
  const int sb0  = (w >> 2) << 1;

  if (t < 64) lred[t] = 0.0f;

  v8f acc0 = {};
  v8f acc1 = {};

  const unsigned short* Win = Wi + (size_t)n * K * SDIM;

  for (int kc = 0; kc < K; kc += 64) {
    __syncthreads();                       // LDS reuse fence
    // stage: 512 granules x 16B each for A and B (pure async copies)
    #pragma unroll
    for (int i = 0; i < 2; ++i) {
      int g  = t + (i << 8);               // 0..511
      int r  = g >> 3;                     // row in 64x64 chunk
      int c8 = (g & 7) << 3;               // 8-aligned column base
      cp16(X + (size_t)(bb + r) * K + kc + c8, &lA[a_frag_off(r, c8)]);
      cp16(Win + (size_t)(kc + r) * SDIM + c8, &lB[b_frag_off(r, c8)]);
    }
    wait_cp();
    __syncthreads();
    #pragma unroll
    for (int kb = 0; kb < 2; ++kb) {
      v16bf av = *(const v16bf*)&lA[(((mblk << 1) + kb) * 32 + lane) * 16];
      v16bf b0, b1;
      #pragma unroll
      for (int s = 0; s < 16; ++s) {
        int kr = ((s >> 3) << 4) + ((lane >> 4) << 3) + (s & 7);
        b0[s] = __builtin_bit_cast(__bf16, lB[(((kb << 2) + sb0)     * 32 + kr) * 16 + (lane & 15)]);
        b1[s] = __builtin_bit_cast(__bf16, lB[(((kb << 2) + sb0 + 1) * 32 + kr) * 16 + (lane & 15)]);
      }
      acc0 = __builtin_amdgcn_wmma_f32_16x16x32_bf16(false, av, false, b0,
                                                     (short)0, acc0, false, false);
      acc1 = __builtin_amdgcn_wmma_f32_16x16x32_bf16(false, av, false, b1,
                                                     (short)0, acc1, false, false);
    }
  }
  __syncthreads();

  // epilogue: bias + liquid activation + Wo dot, S-reduction via LDS atomics
  const int rowloc = (mblk << 4) + ((lane >> 4) << 3);
  #pragma unroll
  for (int j = 0; j < 2; ++j) {
    v8f acc = j ? acc1 : acc0;
    int s = ((sb0 + j) << 4) + (lane & 15);
    float Wos = Wo[n * SDIM + s];
    float bis = bi[n * SDIM + s] + bl[n * SDIM + s];
    #pragma unroll
    for (int v = 0; v < 8; ++v) {
      float li = acc[v] + bis;
      atomicAdd(&lred[rowloc + v], liquid_act(li) * Wos);
    }
  }
  __syncthreads();
  if (t < 64) P[(size_t)(bb + t) * Nn + n] = f2bf(lred[t] + bo[n]);
}

// ---------------------------------------------------------------------------
// Kernel 2: lateral mixing  H = (P + 0.1 * P @ L) * a   (WMMA GEMM, K = Nn)
//   grid = (B/64, Nn/64), block 256.
// ---------------------------------------------------------------------------
__global__ __launch_bounds__(256)
void lateral_kernel(const unsigned short* __restrict__ P,  // [B,Nn] bf16
                    const unsigned short* __restrict__ L,  // [Nn,Nn] bf16
                    const float* __restrict__ a,           // [Nn]
                    unsigned short* __restrict__ H,        // [B,Nn] bf16
                    int Nn)
{
  __shared__ __align__(32) unsigned short lA[4096];
  __shared__ __align__(32) unsigned short lB[4096];

  const int t    = threadIdx.x;
  const int bb   = blockIdx.x * 64;
  const int nb   = blockIdx.y * 64;
  const int w    = t >> 5;
  const int lane = t & 31;
  const int mblk = w & 3;
  const int sb0  = (w >> 2) << 1;

  v8f acc0 = {};
  v8f acc1 = {};

  for (int kc = 0; kc < Nn; kc += 64) {
    __syncthreads();
    #pragma unroll
    for (int i = 0; i < 2; ++i) {
      int g  = t + (i << 8);
      int r  = g >> 3;
      int c8 = (g & 7) << 3;
      cp16(P + (size_t)(bb + r) * Nn + kc + c8, &lA[a_frag_off(r, c8)]);
      cp16(L + (size_t)(kc + r) * Nn + nb + c8, &lB[b_frag_off(r, c8)]);
    }
    wait_cp();
    __syncthreads();
    #pragma unroll
    for (int kb = 0; kb < 2; ++kb) {
      v16bf av = *(const v16bf*)&lA[(((mblk << 1) + kb) * 32 + lane) * 16];
      v16bf b0, b1;
      #pragma unroll
      for (int s = 0; s < 16; ++s) {
        int kr = ((s >> 3) << 4) + ((lane >> 4) << 3) + (s & 7);
        b0[s] = __builtin_bit_cast(__bf16, lB[(((kb << 2) + sb0)     * 32 + kr) * 16 + (lane & 15)]);
        b1[s] = __builtin_bit_cast(__bf16, lB[(((kb << 2) + sb0 + 1) * 32 + kr) * 16 + (lane & 15)]);
      }
      acc0 = __builtin_amdgcn_wmma_f32_16x16x32_bf16(false, av, false, b0,
                                                     (short)0, acc0, false, false);
      acc1 = __builtin_amdgcn_wmma_f32_16x16x32_bf16(false, av, false, b1,
                                                     (short)0, acc1, false, false);
    }
  }

  const int rowloc = (mblk << 4) + ((lane >> 4) << 3);
  #pragma unroll
  for (int j = 0; j < 2; ++j) {
    v8f acc = j ? acc1 : acc0;
    int col = nb + ((sb0 + j) << 4) + (lane & 15);
    float av = a[col];
    #pragma unroll
    for (int v = 0; v < 8; ++v) {
      int row = bb + rowloc + v;
      size_t idx = (size_t)row * Nn + col;
      H[idx] = f2bf((bf2f(P[idx]) + 0.1f * acc[v]) * av);
    }
  }
}

// ---------------------------------------------------------------------------
// Kernel 3: final projection  out[b,o] = H2[b,:] . Wout[:,o] + bout[o]
// ---------------------------------------------------------------------------
__global__ void final_proj_kernel(const unsigned short* __restrict__ H, // [B,64]
                                  const float* __restrict__ Wout,       // [64,12]
                                  const float* __restrict__ bout,       // [12]
                                  float* __restrict__ out)              // [B,12]
{
  int idx = blockIdx.x * blockDim.x + threadIdx.x;
  if (idx >= BATCH * OUTD) return;
  int b = idx / OUTD, o = idx % OUTD;
  float s = bout[o];
  #pragma unroll
  for (int k = 0; k < 64; ++k) s += bf2f(H[b * 64 + k]) * Wout[k * OUTD + o];
  out[idx] = s;
}

// ---------------------------------------------------------------------------
extern "C" void kernel_launch(void* const* d_in, const int* in_sizes, int n_in,
                              void* d_out, int out_size, void* d_ws, size_t ws_size,
                              hipStream_t stream)
{
  const float* x = (const float*)d_in[0];
  const float *Wi[3], *bi[3], *bl[3], *Wo[3], *bo[3], *Lm[3], *ad[3];
  for (int l = 0; l < 3; ++l) {
    int b = 1 + 7 * l;
    Wi[l] = (const float*)d_in[b + 0];
    bi[l] = (const float*)d_in[b + 1];
    bl[l] = (const float*)d_in[b + 2];
    Wo[l] = (const float*)d_in[b + 3];
    bo[l] = (const float*)d_in[b + 4];
    Lm[l] = (const float*)d_in[b + 5];
    ad[l] = (const float*)d_in[b + 6];
  }
  const float* Wout = (const float*)d_in[22];
  const float* bout = (const float*)d_in[23];
  float* out = (float*)d_out;

  // bf16 workspace layout (~33.7 MB)
  unsigned short* us  = (unsigned short*)d_ws;
  unsigned short* Xb  = us;                            // 4096*512
  unsigned short* W0b = Xb  + (size_t)BATCH * 512;     // 256*512*64
  unsigned short* W1b = W0b + (size_t)256 * 512 * 64;  // 128*256*64
  unsigned short* W2b = W1b + (size_t)128 * 256 * 64;  // 64*128*64
  unsigned short* L0b = W2b + (size_t)64 * 128 * 64;   // 256*256
  unsigned short* L1b = L0b + 256 * 256;               // 128*128
  unsigned short* L2b = L1b + 128 * 128;               // 64*64
  unsigned short* P0  = L2b + 64 * 64;                 // 4096*256
  unsigned short* H0  = P0 + (size_t)BATCH * 256;
  unsigned short* P1  = H0 + (size_t)BATCH * 256;      // 4096*128
  unsigned short* H1  = P1 + (size_t)BATCH * 128;
  unsigned short* P2  = H1 + (size_t)BATCH * 128;      // 4096*64
  unsigned short* H2  = P2 + (size_t)BATCH * 64;

  dim3 blk(256);
  auto cvt = [&](const float* s, unsigned short* d, size_t n) {
    int n4 = (int)(n / 4);
    cvt_bf16_kernel<<<dim3((n4 + 255) / 256), blk, 0, stream>>>(s, d, n4);
  };
  cvt(x,     Xb,  (size_t)BATCH * 512);
  cvt(Wi[0], W0b, (size_t)256 * 512 * 64);
  cvt(Wi[1], W1b, (size_t)128 * 256 * 64);
  cvt(Wi[2], W2b, (size_t)64 * 128 * 64);
  cvt(Lm[0], L0b, 256 * 256);
  cvt(Lm[1], L1b, 128 * 128);
  cvt(Lm[2], L2b, 64 * 64);

  liquid_layer_kernel<<<dim3(BATCH / 64, 256), blk, 0, stream>>>(
      Xb, W0b, bi[0], bl[0], Wo[0], bo[0], P0, 512, 256);
  lateral_kernel<<<dim3(BATCH / 64, 4), blk, 0, stream>>>(P0, L0b, ad[0], H0, 256);

  liquid_layer_kernel<<<dim3(BATCH / 64, 128), blk, 0, stream>>>(
      H0, W1b, bi[1], bl[1], Wo[1], bo[1], P1, 256, 128);
  lateral_kernel<<<dim3(BATCH / 64, 2), blk, 0, stream>>>(P1, L1b, ad[1], H1, 128);

  liquid_layer_kernel<<<dim3(BATCH / 64, 64), blk, 0, stream>>>(
      H1, W2b, bi[2], bl[2], Wo[2], bo[2], P2, 128, 64);
  lateral_kernel<<<dim3(BATCH / 64, 1), blk, 0, stream>>>(P2, L2b, ad[2], H2, 64);

  final_proj_kernel<<<dim3((BATCH * OUTD + 255) / 256), blk, 0, stream>>>(
      H2, Wout, bout, out);
}
